// Attention_3590592660193
// MI455X (gfx1250) — compile-verified, matrix operand order checked by reading, not run on hardware
//
#include <hip/hip_runtime.h>
#include <hip/hip_bf16.h>

// ---------------------------------------------------------------------------
// Problem constants (from reference): B=2, S=2048, D=2048, H=16, HD=128
// ---------------------------------------------------------------------------
constexpr int cB  = 2;
constexpr int cS  = 2048;
constexpr int cD  = 2048;
constexpr int cH  = 16;
constexpr int cHD = 128;
constexpr int cM  = cB * cS;     // 4096 rows for the projections

typedef __bf16 bf16_t;
typedef bf16_t v16bf __attribute__((ext_vector_type(16)));
typedef bf16_t v8bf  __attribute__((ext_vector_type(8)));
typedef float  v8f   __attribute__((ext_vector_type(8)));

__device__ __forceinline__ v8f zero8() {
    v8f z;
#pragma unroll
    for (int i = 0; i < 8; ++i) z[i] = 0.0f;
    return z;
}

__device__ __forceinline__ v16bf combine16(v8bf lo, v8bf hi) {
    v16bf r;
#pragma unroll
    for (int i = 0; i < 8; ++i) { r[i] = lo[i]; r[i + 8] = hi[i]; }
    return r;
}

// D = A(16x32 bf16) * B(32x16 bf16) + C (f32), wave32 WMMA
__device__ __forceinline__ v8f wmma_bf16(v16bf a, v16bf b, v8f c) {
    return __builtin_amdgcn_wmma_f32_16x16x32_bf16(
        /*neg_a=*/false, a, /*neg_b=*/false, b,
        /*c_mod=*/(short)0, c, /*reuse_a=*/false, /*reuse_b=*/false);
}

// LDS byte offset of a generic pointer into a __shared__ object (AS3 cast).
__device__ __forceinline__ unsigned lds_off(const void* p) {
    return (unsigned)(size_t)(__attribute__((address_space(3))) const void*)p;
}

// CDNA5 async copy: 16 bytes per lane, global -> LDS, tracked by ASYNCcnt.
__device__ __forceinline__ void async_copy_b128(unsigned ldsaddr, const void* g) {
    asm volatile("global_load_async_to_lds_b128 %0, %1, off"
                 :: "v"(ldsaddr), "v"((unsigned long long)(size_t)g)
                 : "memory");
}

// ---------------------------------------------------------------------------
// Conversion kernels
// ---------------------------------------------------------------------------
__global__ void cvt_f32_to_bf16(const float* __restrict__ in,
                                bf16_t* __restrict__ out, long long n) {
    long long i = (long long)blockIdx.x * blockDim.x + threadIdx.x;
    if (i < n) out[i] = (bf16_t)in[i];
}

// in: fp32 [K][N] row-major  ->  out: bf16 [N][K] row-major (transposed)
__global__ void cvt_transpose_bf16(const float* __restrict__ in,
                                   bf16_t* __restrict__ out, int K, int N) {
    long long i = (long long)blockIdx.x * blockDim.x + threadIdx.x;
    long long total = (long long)K * N;
    if (i >= total) return;
    int n = (int)(i / K);
    int k = (int)(i % K);
    out[i] = (bf16_t)in[(long long)k * N + n];
}

// ---------------------------------------------------------------------------
// Tiled bf16 WMMA GEMM: C[M,N] = A[M,K] * Bt[N,K]^T + bias[N]
// Block tile 128x256, BLK_K=32, 8 waves (2x4), wave tile 64x64 (16 WMMA/step).
// Staging: double-buffered GLOBAL_LOAD_ASYNC_TO_LDS_B128 (ASYNCcnt).
// MODE 0: plain fp32 output
// MODE 1: QKV scatter epilogue -> Q[B,H,S,HD], K[B,H,S,HD], Vt[B,H,HD,S] (bf16)
// ---------------------------------------------------------------------------
constexpr int LDSS = 40;                       // padded LDS row stride (elems)
constexpr unsigned ABUF = 128 * LDSS * 2;      // bytes per A buffer
constexpr unsigned BBUF = 256 * LDSS * 2;      // bytes per B buffer

template <int MODE>
__global__ __launch_bounds__(256)
void gemm_bf16_kernel(const bf16_t* __restrict__ A,
                      const bf16_t* __restrict__ Bt,
                      const float*  __restrict__ bias,
                      float*        __restrict__ Cout,
                      bf16_t*       __restrict__ Qo,
                      bf16_t*       __restrict__ Ko,
                      bf16_t*       __restrict__ Vt,
                      int M, int N, int K) {
    __shared__ __align__(16) bf16_t lA[2][128 * LDSS];
    __shared__ __align__(16) bf16_t lB[2][256 * LDSS];

    const int tid  = threadIdx.x;
    const int wave = tid >> 5;        // 0..7
    const int lane = tid & 31;
    const int half = lane >> 4;       // 0/1
    const int m16  = lane & 15;
    const int wm   = wave >> 2;       // 0..1 -> 64 rows each
    const int wn   = wave & 3;        // 0..3 -> 64 cols each

    const int bm = (int)blockIdx.x * 128;
    const int bn = (int)blockIdx.y * 256;

    v8f acc[4][4];
#pragma unroll
    for (int i = 0; i < 4; ++i)
#pragma unroll
        for (int j = 0; j < 4; ++j) acc[i][j] = zero8();

    // ---- async staging addressing: 6 b128 per thread per tile pair ----
    const int rA = tid >> 1;
    const int eA = (tid & 1) * 16;
    const bf16_t* gA = A  + (size_t)(bm + rA) * K + eA;
    const bf16_t* gB = Bt + (size_t)(bn + tid) * K;
    const unsigned ldsA0 = lds_off(&lA[0][0]) + (unsigned)((rA * LDSS + eA) * 2);
    const unsigned ldsB0 = lds_off(&lB[0][0]) + (unsigned)(tid * LDSS * 2);

    auto issue_tile = [&](int buf, int k0) {
        const unsigned la = ldsA0 + (unsigned)buf * ABUF;
        const unsigned lb = ldsB0 + (unsigned)buf * BBUF;
        async_copy_b128(la,      gA + k0);
        async_copy_b128(la + 16, gA + k0 + 8);
#pragma unroll
        for (int s = 0; s < 4; ++s)
            async_copy_b128(lb + s * 16, gB + k0 + s * 8);
    };

    const int iters = K / 32;
    issue_tile(0, 0);

    for (int i = 0; i < iters; ++i) {
        const int cur = i & 1;
        if (i + 1 < iters) {
            issue_tile(1 - cur, (i + 1) * 32);
            if (i + 2 < iters) {                // warm L2 for tile after next
                __builtin_prefetch(gA + (i + 2) * 32, 0, 1);
                __builtin_prefetch(gB + (i + 2) * 32, 0, 1);
            }
            asm volatile("s_wait_asynccnt 0x6" ::: "memory");
        } else {
            asm volatile("s_wait_asynccnt 0x0" ::: "memory");
        }
        __syncthreads();   // all waves' current tile resident in LDS

        // ---- fragments + 16 WMMAs ----
        v16bf af[4], bfg[4];
        const bf16_t* bufA = &lA[cur][0];
        const bf16_t* bufB = &lB[cur][0];
#pragma unroll
        for (int t = 0; t < 4; ++t) {
            int row = wm * 64 + t * 16 + m16;
            v8bf lo = *(const v8bf*)(bufA + row * LDSS + half * 8);
            v8bf hi = *(const v8bf*)(bufA + row * LDSS + 16 + half * 8);
            af[t] = combine16(lo, hi);
        }
#pragma unroll
        for (int t = 0; t < 4; ++t) {
            int row = wn * 64 + t * 16 + m16;
            v8bf lo = *(const v8bf*)(bufB + row * LDSS + half * 8);
            v8bf hi = *(const v8bf*)(bufB + row * LDSS + 16 + half * 8);
            bfg[t] = combine16(lo, hi);
        }
#pragma unroll
        for (int t = 0; t < 4; ++t)
#pragma unroll
            for (int u = 0; u < 4; ++u)
                acc[t][u] = wmma_bf16(af[t], bfg[u], acc[t][u]);

        __syncthreads();   // protect buffer against rewrite next iteration
    }

    // ---- epilogue ----
#pragma unroll
    for (int mt = 0; mt < 4; ++mt) {
#pragma unroll
        for (int nt = 0; nt < 4; ++nt) {
#pragma unroll
            for (int r = 0; r < 8; ++r) {
                int grow = bm + wm * 64 + mt * 16 + r + half * 8;
                int gcol = bn + wn * 64 + nt * 16 + m16;
                float v = acc[mt][nt][r] + bias[gcol];
                if constexpr (MODE == 0) {
                    Cout[(size_t)grow * N + gcol] = v;
                } else {
                    int h  = gcol / (3 * cHD);
                    int w  = gcol % (3 * cHD);
                    int ty = w >> 7;          // 0:q 1:k 2:v
                    int hd = w & (cHD - 1);
                    int b  = grow / cS;
                    int s  = grow % cS;
                    bf16_t bv = (bf16_t)v;
                    if (ty == 0) {
                        Qo[((size_t)(b * cH + h) * cS + s) * cHD + hd] = bv;
                    } else if (ty == 1) {
                        Ko[((size_t)(b * cH + h) * cS + s) * cHD + hd] = bv;
                    } else {
                        Vt[((size_t)(b * cH + h) * cHD + hd) * cS + s] = bv;
                    }
                }
            }
        }
    }
}

// ---------------------------------------------------------------------------
// Fused causal flash attention (bf16 WMMA, fp32 online softmax)
// grid.x = B*H, grid.y = S/128; block = 256 (8 waves), wave owns 16 q rows.
// K/V staged cooperatively in LDS with double-buffered async copies;
// uniform key-block trip count per block (compute predicated per wave).
// ---------------------------------------------------------------------------
constexpr int KSTR = 136;   // lK row stride elems (128 + 8 pad, 272B)
constexpr int VSTR = 40;    // lV / lP row stride elems (32 + 8 pad, 80B)
constexpr unsigned KBUF = 32  * KSTR * 2;   // bytes per K buffer
constexpr unsigned VBUF = 128 * VSTR * 2;   // bytes per V buffer

__global__ __launch_bounds__(256)
void attn_fwd_kernel(const bf16_t* __restrict__ Q,
                     const bf16_t* __restrict__ Kk,
                     const bf16_t* __restrict__ Vt,
                     bf16_t* __restrict__ Out) {
    __shared__ __align__(16) bf16_t lK[2][32 * KSTR];    // 32 keys x 128 hd
    __shared__ __align__(16) bf16_t lV[2][128 * VSTR];   // 128 hd x 32 keys
    __shared__ __align__(16) bf16_t lP[8][16 * VSTR];    // per-wave P tile

    const int tid  = threadIdx.x;
    const int wave = tid >> 5;
    const int lane = tid & 31;
    const int half = lane >> 4;
    const int m16  = lane & 15;

    const int bh    = (int)blockIdx.x;            // b*H + h
    const int qt    = (int)blockIdx.y;            // 128-row q supertile
    const int qbase = qt * 128 + wave * 16;

    const bf16_t* Qp = Q  + (size_t)bh * cS * cHD;
    const bf16_t* Kp = Kk + (size_t)bh * cS * cHD;
    const bf16_t* Vp = Vt + (size_t)bh * cHD * cS;

    // Q fragments: 16 rows x 128 (4 k-chunks of 32), loaded once from global
    v16bf aq[4];
#pragma unroll
    for (int c = 0; c < 4; ++c) {
        const bf16_t* p = Qp + (size_t)(qbase + m16) * cHD + c * 32 + half * 8;
        aq[c] = combine16(*(const v8bf*)p, *(const v8bf*)(p + 16));
    }

    v8f o[8];
#pragma unroll
    for (int t = 0; t < 8; ++t) o[t] = zero8();
    float rmax[8], rsum[8];
#pragma unroll
    for (int r = 0; r < 8; ++r) { rmax[r] = -3.0e38f; rsum[r] = 0.0f; }

    const float scale = 0.08838834764831845f;     // 1/sqrt(128)

    // cooperative K/V staging: 4 async b128 per thread per 32-key block
    auto issue_kv = [&](int buf, int kb) {
#pragma unroll
        for (int s = 0; s < 2; ++s) {
            int seg = tid * 2 + s;                 // 0..511
            int rk = seg >> 4, ck = seg & 15;      // K: 32 rows x 16 segs
            async_copy_b128(
                lds_off(&lK[buf][0]) + (unsigned)((rk * KSTR + ck * 8) * 2),
                Kp + (size_t)(kb + rk) * cHD + ck * 8);
            int rv = seg >> 2, cv = seg & 3;       // V: 128 rows x 4 segs
            async_copy_b128(
                lds_off(&lV[buf][0]) + (unsigned)((rv * VSTR + cv * 8) * 2),
                Vp + (size_t)rv * cS + kb + cv * 8);
        }
    };

    const int nblk = qt * 4 + 4;                   // uniform across the block
    issue_kv(0, 0);

    for (int ib = 0; ib < nblk; ++ib) {
        const int cur = ib & 1;
        const int kb  = ib * 32;
        if (ib + 1 < nblk) {
            issue_kv(1 - cur, kb + 32);
            asm volatile("s_wait_asynccnt 0x4" ::: "memory");
        } else {
            asm volatile("s_wait_asynccnt 0x0" ::: "memory");
        }
        __syncthreads();   // current K/V tile resident for all waves

        if (kb <= qbase + 15) {   // wave-uniform causal predicate
            // ---- logits: Q[16x128] x K^T[128x32] via LDS ----
            v8f s0 = zero8(), s1 = zero8();
            const bf16_t* kbuf = &lK[cur][0];
#pragma unroll
            for (int c = 0; c < 4; ++c) {
                const bf16_t* kp0 = kbuf + m16 * KSTR + c * 32 + half * 8;
                const bf16_t* kp1 = kbuf + (16 + m16) * KSTR + c * 32 + half * 8;
                v16bf bk0 = combine16(*(const v8bf*)kp0, *(const v8bf*)(kp0 + 16));
                v16bf bk1 = combine16(*(const v8bf*)kp1, *(const v8bf*)(kp1 + 16));
                s0 = wmma_bf16(aq[c], bk0, s0);
                s1 = wmma_bf16(aq[c], bk1, s1);
            }

            // ---- causal mask + online softmax (half-wave reductions) ----
            const int col0 = kb + m16;
            const int col1 = kb + 16 + m16;
            bf16_t* pb = &lP[wave][0];
            float corr[8];
#pragma unroll
            for (int r = 0; r < 8; ++r) {
                int row = qbase + r + half * 8;
                float l0 = (col0 <= row) ? s0[r] * scale : -3.0e38f;
                float l1 = (col1 <= row) ? s1[r] * scale : -3.0e38f;
                float mx = fmaxf(l0, l1);
#pragma unroll
                for (int off = 1; off < 16; off <<= 1)
                    mx = fmaxf(mx, __shfl_xor(mx, off, 16));
                float nm = fmaxf(rmax[r], mx);
                float cf = __expf(rmax[r] - nm);
                rmax[r]  = nm;
                float p0 = __expf(l0 - nm);
                float p1 = __expf(l1 - nm);
                float ps = p0 + p1;
#pragma unroll
                for (int off = 1; off < 16; off <<= 1)
                    ps += __shfl_xor(ps, off, 16);
                rsum[r] = rsum[r] * cf + ps;
                corr[r] = cf;
                pb[(r + half * 8) * VSTR + m16]      = (bf16_t)p0;
                pb[(r + half * 8) * VSTR + 16 + m16] = (bf16_t)p1;
            }
#pragma unroll
            for (int t = 0; t < 8; ++t)
#pragma unroll
                for (int r = 0; r < 8; ++r) o[t][r] *= corr[r];

            // wave-synchronous LDS visibility for the P tile
            asm volatile("s_wait_dscnt 0" ::: "memory");

            // ---- P as A-fragment (16 rows x 32 keys) ----
            const bf16_t* pr = &lP[wave][m16 * VSTR + half * 8];
            v16bf ap = combine16(*(const v8bf*)pr, *(const v8bf*)(pr + 16));

            // ---- o += P x V from staged LDS ----
            const bf16_t* vbuf = &lV[cur][0];
#pragma unroll
            for (int t = 0; t < 8; ++t) {
                const bf16_t* vp = vbuf + (t * 16 + m16) * VSTR + half * 8;
                v16bf bv = combine16(*(const v8bf*)vp, *(const v8bf*)(vp + 16));
                o[t] = wmma_bf16(ap, bv, o[t]);
            }
        }

        __syncthreads();   // all waves done with this buffer before rewrite
    }

    // ---- normalize and write attn (bf16 [B,S,D]) ----
    const int b = bh / cH;
    const int h = bh % cH;
#pragma unroll
    for (int t = 0; t < 8; ++t) {
#pragma unroll
        for (int r = 0; r < 8; ++r) {
            int row = qbase + r + half * 8;
            float val = o[t][r] / rsum[r];
            Out[((size_t)(b * cS + row)) * cD + h * cHD + t * 16 + m16] = (bf16_t)val;
        }
    }
}

// ---------------------------------------------------------------------------
// Launch
// ---------------------------------------------------------------------------
extern "C" void kernel_launch(void* const* d_in, const int* in_sizes, int n_in,
                              void* d_out, int out_size, void* d_ws, size_t ws_size,
                              hipStream_t stream) {
    const float* x    = (const float*)d_in[0];   // [B,S,D]
    const float* Wqkv = (const float*)d_in[1];   // [D,3D]
    const float* bqkv = (const float*)d_in[2];   // [3D]
    const float* Wo   = (const float*)d_in[3];   // [D,D]
    const float* bo   = (const float*)d_in[4];   // [D]
    float* out = (float*)d_out;

    const size_t nBSD = (size_t)cB * cS * cD;          // 8,388,608
    const size_t nQKV = (size_t)cD * 3 * cD;           // 12,582,912
    const size_t nWo  = (size_t)cD * cD;               // 4,194,304

    // workspace carve-up (bf16 buffers), 256B aligned
    char* ws = (char*)d_ws;
    size_t off = 0;
    auto carve = [&](size_t elems) {
        char* p = ws + off;
        off += (elems * sizeof(bf16_t) + 255) & ~(size_t)255;
        return (bf16_t*)p;
    };
    bf16_t* xbf   = carve(nBSD);
    bf16_t* WqkvT = carve(nQKV);
    bf16_t* WoT   = carve(nWo);
    bf16_t* Qw    = carve(nBSD);
    bf16_t* Kw    = carve(nBSD);
    bf16_t* Vtw   = carve(nBSD);
    bf16_t* attnw = carve(nBSD);

    // 1) conversions / weight transposes
    cvt_f32_to_bf16<<<(unsigned)((nBSD + 255) / 256), 256, 0, stream>>>(
        x, xbf, (long long)nBSD);
    cvt_transpose_bf16<<<(unsigned)((nQKV + 255) / 256), 256, 0, stream>>>(
        Wqkv, WqkvT, cD, 3 * cD);
    cvt_transpose_bf16<<<(unsigned)((nWo + 255) / 256), 256, 0, stream>>>(
        Wo, WoT, cD, cD);

    // 2) fused QKV projection -> Q, K, V^T (bf16)
    {
        dim3 grid(cM / 128, (3 * cD) / 256);
        gemm_bf16_kernel<1><<<grid, 256, 0, stream>>>(
            xbf, WqkvT, bqkv, nullptr, Qw, Kw, Vtw, cM, 3 * cD, cD);
    }

    // 3) fused causal attention
    {
        dim3 grid(cB * cH, cS / 128);
        attn_fwd_kernel<<<grid, 256, 0, stream>>>(Qw, Kw, Vtw, attnw);
    }

    // 4) output projection -> fp32 d_out
    {
        dim3 grid(cM / 128, cD / 256);
        gemm_bf16_kernel<0><<<grid, 256, 0, stream>>>(
            attnw, WoT, bo, out, nullptr, nullptr, nullptr, cM, cD, cD);
    }
}
